// ExpertMLP_17961553232371
// MI455X (gfx1250) — compile-verified, matrix operand order checked by reading, not run on hardware
//
#include <hip/hip_runtime.h>
#include <hip/hip_bf16.h>
#include <math.h>

typedef __attribute__((ext_vector_type(16))) _Float16 v16h;
typedef __attribute__((ext_vector_type(8)))  _Float16 v8h;
typedef __attribute__((ext_vector_type(8)))  float    v8f;

#define NE    8
#define DIN   512
#define DHID  512
#define DOUT  512
#define NB    1024
#define KDIM  (NE * DIN)      // 4096 (same for both layers)
#define BN_EPS 1e-5f

// ---------------------------------------------------------------------------
// 1) X'[b, e*DIN + i] = (f16)(coef[b,e] * x[b,i])
// ---------------------------------------------------------------------------
__global__ __launch_bounds__(256)
void expand_x_kernel(const float* __restrict__ x, const float* __restrict__ coef,
                     _Float16* __restrict__ Xp) {
    int t = blockIdx.x * blockDim.x + threadIdx.x;     // over NB*DIN
    int b = t / DIN, i = t - b * DIN;
    float xv = x[(size_t)b * DIN + i];
    const float* c = coef + (size_t)b * NE;
    _Float16* o = Xp + (size_t)b * KDIM + i;
#pragma unroll
    for (int e = 0; e < NE; ++e)
        o[(size_t)e * DIN] = (_Float16)(c[e] * xv);
}

// ---------------------------------------------------------------------------
// 2) LDS-tiled transpose + f32->f16 convert: W[K,N] -> Wt[N,K]
//    (makes the WMMA B-fragment two contiguous 16B loads per lane)
// ---------------------------------------------------------------------------
__global__ __launch_bounds__(256)
void transpose_convert_kernel(const float* __restrict__ W, _Float16* __restrict__ Wt,
                              int K, int N) {
    __shared__ float tile[16][17];
    int k0 = blockIdx.x * 16, n0 = blockIdx.y * 16;
    int tx = threadIdx.x, ty = threadIdx.y;
    tile[ty][tx] = W[(size_t)(k0 + ty) * N + (n0 + tx)];
    __syncthreads();
    Wt[(size_t)(n0 + ty) * K + (k0 + tx)] = (_Float16)tile[tx][ty];
}

// ---------------------------------------------------------------------------
// Fragment loader per the CDNA5 16-bit 16x32 A/B VGPR layout:
//   elements 0..7  <- p[0..7]       (this lane's first K-octet)
//   elements 8..15 <- p[16..23]     (same octet position, +16 in K)
// p already includes k0 + (lane>>4)*8.
// ---------------------------------------------------------------------------
__device__ __forceinline__ v16h load_frag16(const _Float16* __restrict__ p) {
    v8h lo = *(const v8h*)(p);
    v8h hi = *(const v8h*)(p + 16);
    v16h r;
#pragma unroll
    for (int i = 0; i < 8; ++i) { r[i] = lo[i]; r[i + 8] = hi[i]; }
    return r;
}

// ---------------------------------------------------------------------------
// 3) GEMM: out[M,N] = A[M,K](f16) @ Bt[N,K]^T(f16)  + coef @ biasEN
//    One wave computes a 16(M) x 64(N) tile: A-frag reused across 4 WMMAs.
//    M=1024, N=512, K=4096.  512 waves -> 64 blocks of 256 threads.
// ---------------------------------------------------------------------------
__global__ __launch_bounds__(256)
void gemm_wmma_kernel(const _Float16* __restrict__ A,
                      const _Float16* __restrict__ Bt,
                      const float* __restrict__ coef,     // [M, NE]
                      const float* __restrict__ biasEN,   // [NE, N]
                      float* __restrict__ out,            // [M, N] f32
                      int M, int N, int K) {
    const int wave = blockIdx.x * (blockDim.x >> 5) + (threadIdx.x >> 5);
    const int lane = threadIdx.x & 31;
    const int nTiles = N >> 6;                  // tiles of 64 cols
    const int tm = (wave / nTiles) << 4;
    const int tn = (wave % nTiles) << 6;
    const int lr = lane & 15;
    const int kg = lane >> 4;                   // K-octet group select

    const _Float16* arow = A  + (size_t)(tm + lr) * K + kg * 8;
    const _Float16* brow = Bt + (size_t)(tn + lr) * K + kg * 8;

    v8f acc0 = {}, acc1 = {}, acc2 = {}, acc3 = {};

    for (int k0 = 0; k0 < K; k0 += 32) {
        v16h a  = load_frag16(arow + k0);
        v16h b0 = load_frag16(brow + k0);
        v16h b1 = load_frag16(brow + (size_t)16 * K + k0);
        v16h b2 = load_frag16(brow + (size_t)32 * K + k0);
        v16h b3 = load_frag16(brow + (size_t)48 * K + k0);
        acc0 = __builtin_amdgcn_wmma_f32_16x16x32_f16(false, a, false, b0, (short)0, acc0, false, false);
        acc1 = __builtin_amdgcn_wmma_f32_16x16x32_f16(false, a, false, b1, (short)0, acc1, false, false);
        acc2 = __builtin_amdgcn_wmma_f32_16x16x32_f16(false, a, false, b2, (short)0, acc2, false, false);
        acc3 = __builtin_amdgcn_wmma_f32_16x16x32_f16(false, a, false, b3, (short)0, acc3, false, false);
    }

    // Epilogue. C/D layout: lane(0..15)=N col, VGPR r -> M row r (lanes 0-15)
    // or r+8 (lanes 16-31).
    float bcol[4][NE];
#pragma unroll
    for (int j = 0; j < 4; ++j) {
        int n = tn + j * 16 + lr;
#pragma unroll
        for (int e = 0; e < NE; ++e) bcol[j][e] = biasEN[(size_t)e * N + n];
    }
    const int rowbase = tm + kg * 8;
#pragma unroll
    for (int r = 0; r < 8; ++r) {
        const int row = rowbase + r;
        const float* cr = coef + (size_t)row * NE;
        float cf[NE];
#pragma unroll
        for (int e = 0; e < NE; ++e) cf[e] = cr[e];
#pragma unroll
        for (int j = 0; j < 4; ++j) {
            float bv = 0.f;
#pragma unroll
            for (int e = 0; e < NE; ++e) bv += cf[e] * bcol[j][e];
            float cv = (j == 0) ? acc0[r] : (j == 1) ? acc1[r] : (j == 2) ? acc2[r] : acc3[r];
            out[(size_t)row * N + tn + j * 16 + lr] = cv + bv;
        }
    }
}

// ---------------------------------------------------------------------------
// 4) BatchNorm stats per hidden unit (biased variance), fused to scale/shift
// ---------------------------------------------------------------------------
__global__ __launch_bounds__(256)
void bn_stats_kernel(const float* __restrict__ h, const float* __restrict__ gamma,
                     const float* __restrict__ beta, float* __restrict__ scale,
                     float* __restrict__ shift) {
    const int n = blockIdx.x;                  // hidden unit
    float s = 0.f, s2 = 0.f;
    for (int r = threadIdx.x; r < NB; r += blockDim.x) {
        float v = h[(size_t)r * DHID + n];
        s += v; s2 += v * v;
    }
    __shared__ float sh[256], sh2[256];
    sh[threadIdx.x] = s; sh2[threadIdx.x] = s2;
    __syncthreads();
    for (int off = 128; off > 0; off >>= 1) {
        if (threadIdx.x < off) {
            sh[threadIdx.x]  += sh[threadIdx.x + off];
            sh2[threadIdx.x] += sh2[threadIdx.x + off];
        }
        __syncthreads();
    }
    if (threadIdx.x == 0) {
        float mean = sh[0] * (1.0f / NB);
        float var  = sh2[0] * (1.0f / NB) - mean * mean;
        float sc   = gamma[n] * rsqrtf(var + BN_EPS);
        scale[n] = sc;
        shift[n] = beta[n] - mean * sc;
    }
}

// ---------------------------------------------------------------------------
// 5) H1'[b, e*DHID + h] = (f16)(coef[b,e] * elu(h*scale + shift))
// ---------------------------------------------------------------------------
__global__ __launch_bounds__(256)
void expand_h_kernel(const float* __restrict__ h, const float* __restrict__ coef,
                     const float* __restrict__ scale, const float* __restrict__ shift,
                     _Float16* __restrict__ Hp) {
    int t = blockIdx.x * blockDim.x + threadIdx.x;     // over NB*DHID
    int b = t / DHID, hh = t - b * DHID;
    float v = h[t] * scale[hh] + shift[hh];
    v = (v > 0.f) ? v : (expf(v) - 1.f);               // ELU, alpha = 1
    const float* c = coef + (size_t)b * NE;
    _Float16* o = Hp + (size_t)b * (NE * DHID) + hh;
#pragma unroll
    for (int e = 0; e < NE; ++e)
        o[(size_t)e * DHID] = (_Float16)(c[e] * v);
}

// ---------------------------------------------------------------------------
// Launch
// ---------------------------------------------------------------------------
extern "C" void kernel_launch(void* const* d_in, const int* in_sizes, int n_in,
                              void* d_out, int out_size, void* d_ws, size_t ws_size,
                              hipStream_t stream) {
    const float* x     = (const float*)d_in[0];   // [NB, DIN]
    const float* coef  = (const float*)d_in[1];   // [NB, NE]
    const float* w1    = (const float*)d_in[2];   // [NE, DIN, DHID] == [KDIM, DHID]
    const float* b1    = (const float*)d_in[3];   // [NE, DHID]
    const float* w2    = (const float*)d_in[4];   // [NE, DHID, DOUT] == [KDIM, DOUT]
    const float* b2    = (const float*)d_in[5];   // [NE, DOUT]
    const float* gamma = (const float*)d_in[6];   // [DHID]
    const float* beta  = (const float*)d_in[7];   // [DHID]
    float* out = (float*)d_out;                   // [NB, DOUT]

    char* ws = (char*)d_ws;
    const size_t MiB = 1024u * 1024u;
    _Float16* Xp  = (_Float16*)(ws + 0);          // 8 MiB  [NB, KDIM]  (reused as Hp)
    _Float16* W1t = (_Float16*)(ws + 8  * MiB);   // 4 MiB  [DHID, KDIM]
    _Float16* W2t = (_Float16*)(ws + 12 * MiB);   // 4 MiB  [DOUT, KDIM]
    float*    hbuf  = (float*)(ws + 16 * MiB);    // 2 MiB  [NB, DHID]
    float*    scale = (float*)(ws + 18 * MiB);    // 2 KiB
    float*    shift = (float*)(ws + 18 * MiB + 4096);

    // 1) expand x with blending coefs
    expand_x_kernel<<<(NB * DIN) / 256, 256, 0, stream>>>(x, coef, Xp);

    // 2) transpose+convert weights (B operand K-contiguous)
    transpose_convert_kernel<<<dim3(KDIM / 16, DHID / 16), dim3(16, 16), 0, stream>>>(w1, W1t, KDIM, DHID);
    transpose_convert_kernel<<<dim3(KDIM / 16, DOUT / 16), dim3(16, 16), 0, stream>>>(w2, W2t, KDIM, DOUT);

    // 3) layer-1 GEMM (WMMA) + bias
    {
        int waves = (NB / 16) * (DHID / 64);      // 512
        gemm_wmma_kernel<<<waves / 8, 256, 0, stream>>>(Xp, W1t, coef, b1, hbuf, NB, DHID, KDIM);
    }

    // 4) batchnorm stats -> fused scale/shift
    bn_stats_kernel<<<DHID, 256, 0, stream>>>(hbuf, gamma, beta, scale, shift);

    // 5) BN + ELU + expand (overwrites Xp, which is dead now)
    expand_h_kernel<<<(NB * DHID) / 256, 256, 0, stream>>>(hbuf, coef, scale, shift, Xp);

    // 6) layer-2 GEMM (WMMA) + bias -> final output
    {
        int waves = (NB / 16) * (DOUT / 64);      // 512
        gemm_wmma_kernel<<<waves / 8, 256, 0, stream>>>(Xp, W2t, coef, b2, out, NB, DOUT, KDIM);
    }
}
